// VanillaRNN_35150012351213
// MI455X (gfx1250) — matrix operand order repaired and checked by reading, not compiled, not run on hardware
//
#include <hip/hip_runtime.h>
#include <hip/hip_bf16.h>
#include <math.h>

typedef __attribute__((ext_vector_type(16))) __bf16 v16bf;
typedef __attribute__((ext_vector_type(8)))  float  v8f;

#define SEQ_LEN   128
#define HID       2048
#define BATCH_N   256
#define NCLS      10

// ---------------------------------------------------------------------------
// Kernel 0: convert W_h2h fp32 -> bf16 (row-major 2048x2048, W[n,k])
// ---------------------------------------------------------------------------
__global__ __launch_bounds__(256)
void rnn_cvt_w_bf16(const float* __restrict__ w, __bf16* __restrict__ wb, int n) {
    int i = (blockIdx.x * blockDim.x + threadIdx.x) * 4;
    if (i < n) {
        float4 f = *(const float4*)(w + i);
        union { uint2 u; __bf16 h[4]; } o;
        o.h[0] = (__bf16)f.x; o.h[1] = (__bf16)f.y;
        o.h[2] = (__bf16)f.z; o.h[3] = (__bf16)f.w;
        *(uint2*)(wb + i) = o.u;
    }
}

// ---------------------------------------------------------------------------
// Kernel 1: zero the bf16 h0 buffer
// ---------------------------------------------------------------------------
__global__ __launch_bounds__(256)
void rnn_zero_bf16(__bf16* __restrict__ p, int n) {
    int i = (blockIdx.x * blockDim.x + threadIdx.x) * 4;
    if (i < n) {
        uint2 z = {0u, 0u};
        *(uint2*)(p + i) = z;
    }
}

// ---------------------------------------------------------------------------
// Kernel 2: one recurrence step.
//   h_out = tanh( x[:,t]*w_i2h + bias_h + h_in @ W^T )
// C = A @ B with A = h_in (256x2048 bf16), B[k,n] = W[n,k] (columns of B are
// contiguous rows of W -> contiguous 32B loads per lane, no transpose needed).
//
// Each wave computes a 16(M) x 64(N) output tile: 4 accumulators, A reused 4x.
// Block = 256 threads = 8 waves that ALL share the same N macro-tile and thus
// issue identical B addresses -> 8-way WGP$ sharing of W.
// The k-loop is explicitly double-buffered (slots a/b) so the loads for step
// k+32 are in flight while the WMMAs for step k execute — avoids the
// s_wait_loadcnt 0x0 full-drain before every WMMA.
// Grid: 64 blocks = 32 N macro-tiles x 2 M groups.
// ---------------------------------------------------------------------------
__global__ __launch_bounds__(256)
void rnn_step_wmma(const __bf16* __restrict__ hin,
                   const __bf16* __restrict__ W,
                   const float*  __restrict__ x,     // (BATCH, SEQ)
                   const float*  __restrict__ wi,    // (HID)
                   const float*  __restrict__ bh,    // (HID)
                   __bf16*       __restrict__ hout,
                   float*        __restrict__ hfin,  // non-null on last step
                   int t)
{
    const int lane   = threadIdx.x & 31;
    const int waveid = threadIdx.x >> 5;          // 0..7
    const int nt     = blockIdx.x & 31;           // 32 N macro-tiles of 64
    const int mt     = ((blockIdx.x >> 5) << 3) + waveid; // 16 M tiles
    const int m0     = mt * 16;
    const int n0     = nt * 64;
    const int l15    = lane & 15;
    const int lhi    = lane >> 4;                 // 0 or 1

    // A-matrix (16x32 bf16): lane l<16 holds row M=l, K {0..15};
    // lane l+16 holds row M=l, K {16..31}  -> two 16B loads per lane.
    const __bf16* aptr = hin + (size_t)(m0 + l15) * HID + (lhi << 4);

    // B-matrix (32x16 bf16): lane holds column N=l15, K span lhi*16..+15,
    // contiguous in W's row -> one 32B load (two uint4) per lane per tile.
    const __bf16* bptr0 = W + (size_t)(n0 +  0 + l15) * HID + (lhi << 4);
    const __bf16* bptr1 = W + (size_t)(n0 + 16 + l15) * HID + (lhi << 4);
    const __bf16* bptr2 = W + (size_t)(n0 + 32 + l15) * HID + (lhi << 4);
    const __bf16* bptr3 = W + (size_t)(n0 + 48 + l15) * HID + (lhi << 4);

    v8f acc0 = {}; v8f acc1 = {}; v8f acc2 = {}; v8f acc3 = {};

    union ABu { v16bf v; uint4 u[2]; };
    ABu Aa, Ab, B0a, B0b, B1a, B1b, B2a, B2b, B3a, B3b;

#define RNN_LOAD(S, KK)                                        \
    A##S.u[0]  = *(const uint4*)(aptr  + (KK));                \
    A##S.u[1]  = *(const uint4*)(aptr  + (KK) + 8);            \
    B0##S.u[0] = *(const uint4*)(bptr0 + (KK));                \
    B0##S.u[1] = *(const uint4*)(bptr0 + (KK) + 8);            \
    B1##S.u[0] = *(const uint4*)(bptr1 + (KK));                \
    B1##S.u[1] = *(const uint4*)(bptr1 + (KK) + 8);            \
    B2##S.u[0] = *(const uint4*)(bptr2 + (KK));                \
    B2##S.u[1] = *(const uint4*)(bptr2 + (KK) + 8);            \
    B3##S.u[0] = *(const uint4*)(bptr3 + (KK));                \
    B3##S.u[1] = *(const uint4*)(bptr3 + (KK) + 8);

#define RNN_MMA(S)                                                             \
    acc0 = __builtin_amdgcn_wmma_f32_16x16x32_bf16(false, A##S.v, false,       \
                 B0##S.v, (short)0, acc0, false, false);                       \
    acc1 = __builtin_amdgcn_wmma_f32_16x16x32_bf16(false, A##S.v, false,       \
                 B1##S.v, (short)0, acc1, false, false);                       \
    acc2 = __builtin_amdgcn_wmma_f32_16x16x32_bf16(false, A##S.v, false,       \
                 B2##S.v, (short)0, acc2, false, false);                       \
    acc3 = __builtin_amdgcn_wmma_f32_16x16x32_bf16(false, A##S.v, false,       \
                 B3##S.v, (short)0, acc3, false, false);

    // Prologue: fill slot a with k-step 0.
    RNN_LOAD(a, 0)
    // Steady state: 2 k-steps per iteration, loads one step ahead of WMMAs.
    for (int kk = 0; kk < HID - 64; kk += 64) {
        RNN_LOAD(b, kk + 32)
        RNN_MMA(a)
        RNN_LOAD(a, kk + 64)
        RNN_MMA(b)
    }
    // Tail: k-steps HID-64 (slot a) and HID-32 (slot b).
    RNN_LOAD(b, HID - 32)
    RNN_MMA(a)
    RNN_MMA(b)

#undef RNN_LOAD
#undef RNN_MMA

    // Epilogue: C/D layout — lane holds column N=l15; VGPR r holds row
    // M = r + 8*lhi. Fold in xin_t = x[b,t]*wi[n] + bh[n], apply tanh.
    const int rbase = m0 + (lhi << 3);
    float xr[8];
    #pragma unroll
    for (int r = 0; r < 8; ++r)
        xr[r] = x[(rbase + r) * SEQ_LEN + t];

    v8f accs[4] = {acc0, acc1, acc2, acc3};
    if (hfin == nullptr) {
        #pragma unroll
        for (int j = 0; j < 4; ++j) {
            const int col = n0 + j * 16 + l15;
            const float wic = wi[col];
            const float bhc = bh[col];
            #pragma unroll
            for (int r = 0; r < 8; ++r) {
                float h = tanhf(accs[j][r] + xr[r] * wic + bhc);
                hout[(size_t)(rbase + r) * HID + col] = (__bf16)h;
            }
        }
    } else {
        #pragma unroll
        for (int j = 0; j < 4; ++j) {
            const int col = n0 + j * 16 + l15;
            const float wic = wi[col];
            const float bhc = bh[col];
            #pragma unroll
            for (int r = 0; r < 8; ++r) {
                float h = tanhf(accs[j][r] + xr[r] * wic + bhc);
                hout[(size_t)(rbase + r) * HID + col] = (__bf16)h;
                hfin[(size_t)(rbase + r) * HID + col] = h;
            }
        }
    }
}

// ---------------------------------------------------------------------------
// Kernel 3: p = h_final @ w_h2p^T + bias_p ; softmax.  One wave per batch row.
// ---------------------------------------------------------------------------
__global__ __launch_bounds__(32)
void rnn_proj_softmax(const float* __restrict__ hfin,  // (BATCH, HID) fp32
                      const float* __restrict__ wp,    // (NCLS, HID)
                      const float* __restrict__ bp,    // (NCLS)
                      float*       __restrict__ out)   // (BATCH, NCLS)
{
    const int b    = blockIdx.x;
    const int lane = threadIdx.x;

    float acc[NCLS];
    #pragma unroll
    for (int c = 0; c < NCLS; ++c) acc[c] = 0.f;

    for (int k = lane; k < HID; k += 32) {
        float hv = hfin[(size_t)b * HID + k];
        #pragma unroll
        for (int c = 0; c < NCLS; ++c)
            acc[c] += hv * wp[(size_t)c * HID + k];
    }
    #pragma unroll
    for (int c = 0; c < NCLS; ++c)
        for (int off = 16; off > 0; off >>= 1)
            acc[c] += __shfl_xor(acc[c], off, 32);

    if (lane == 0) {
        float p[NCLS], mx = -3.0e38f;
        #pragma unroll
        for (int c = 0; c < NCLS; ++c) { p[c] = acc[c] + bp[c]; mx = fmaxf(mx, p[c]); }
        float s = 0.f;
        #pragma unroll
        for (int c = 0; c < NCLS; ++c) { p[c] = __expf(p[c] - mx); s += p[c]; }
        float inv = 1.f / s;
        #pragma unroll
        for (int c = 0; c < NCLS; ++c) out[b * NCLS + c] = p[c] * inv;
    }
}

// ---------------------------------------------------------------------------
extern "C" void kernel_launch(void* const* d_in, const int* in_sizes, int n_in,
                              void* d_out, int out_size, void* d_ws, size_t ws_size,
                              hipStream_t stream) {
    const float* x      = (const float*)d_in[0]; // (256,128)
    const float* w_i2h  = (const float*)d_in[1]; // (2048,1)
    const float* w_h2h  = (const float*)d_in[2]; // (2048,2048)
    const float* w_h2p  = (const float*)d_in[3]; // (10,2048)
    const float* bias_h = (const float*)d_in[4]; // (2048,1)
    const float* bias_p = (const float*)d_in[5]; // (10,1)
    float* out = (float*)d_out;

    // Workspace layout (≈12.6 MB)
    char* ws = (char*)d_ws;
    const size_t W_BYTES = (size_t)HID * HID * sizeof(__bf16);       // 8 MB
    const size_t H_BYTES = (size_t)BATCH_N * HID * sizeof(__bf16);   // 1 MB
    __bf16* Wb   = (__bf16*)ws;
    __bf16* hA   = (__bf16*)(ws + W_BYTES);
    __bf16* hB   = (__bf16*)(ws + W_BYTES + H_BYTES);
    float*  hfin = (float*)(ws + W_BYTES + 2 * H_BYTES);

    // 0) W fp32 -> bf16
    {
        int n = HID * HID;                       // 4,194,304
        int blocks = n / (256 * 4);              // 4096
        rnn_cvt_w_bf16<<<blocks, 256, 0, stream>>>(w_h2h, Wb, n);
    }
    // 1) h0 = 0
    {
        int n = BATCH_N * HID;                   // 524,288
        int blocks = n / (256 * 4);              // 512
        rnn_zero_bf16<<<blocks, 256, 0, stream>>>(hA, n);
    }
    // 2) 128 recurrence steps, ping-pong h buffers.
    __bf16* hin = hA; __bf16* hout = hB;
    for (int t = 0; t < SEQ_LEN; ++t) {
        float* fin = (t == SEQ_LEN - 1) ? hfin : nullptr;
        rnn_step_wmma<<<64, 256, 0, stream>>>(hin, Wb, x, w_i2h, bias_h,
                                              hout, fin, t);
        __bf16* tmp = hin; hin = hout; hout = tmp;
    }
    // 3) projection + softmax
    rnn_proj_softmax<<<BATCH_N, 32, 0, stream>>>(hfin, w_h2p, bias_p, out);
}
